// LESA_66434554134791
// MI455X (gfx1250) — matrix operand order repaired and checked by reading, not compiled, not run on hardware
//
#include <hip/hip_runtime.h>

// ---------------- problem constants ----------------
#define N_    8
#define C_    512
#define HW_   784
#define G_    8
#define QKP_  32
#define VP_   64
#define BP_   512
#define PE_   128          // 2*QKP + VP
#define O2_   1024         // BP*2
#define T_    1567         // 2*HW - 1
#define TPAD_ 1600         // padded rel columns
#define TROWS_ 1568        // padded relT rows (row 1567 == 0)
#define JPAD_ 800          // padded j/t extent (multiple of 32)
#define SRW_  832          // simR row stride (16 front pad + 800 + tail)
#define RS_   0.9999950000374997f   // 1/sqrt(1 + 1e-5)

typedef __attribute__((ext_vector_type(16))) __bf16          v16bf;
typedef __attribute__((ext_vector_type(8)))  float           v8f;
typedef __attribute__((ext_vector_type(8)))  unsigned short  ushort8;

static __device__ __forceinline__ unsigned short f2bf(float f) {
  unsigned int u = __float_as_uint(f);
  unsigned int r = u + 0x7FFFu + ((u >> 16) & 1u);   // RNE truncation
  return (unsigned short)(r >> 16);
}

static __device__ __forceinline__ v8f wmma_bf16(v16bf a, v16bf b, v8f c) {
  return __builtin_amdgcn_wmma_f32_16x16x32_bf16(false, a, false, b, (short)0, c,
                                                 false, false);
}

// A-tile (16x32 bf16): lane<16 holds M=lane, K={0..7,16..23}; lane>=16 M=lane-16, K={8..15,24..31}
static __device__ __forceinline__ v16bf load_bf16_A(const unsigned short* row, int lane) {
  int off1 = (lane < 16) ? 0 : 8;
  union { v16bf v; ushort8 h[2]; } u;
  u.h[0] = *(const ushort8*)(row + off1);
  u.h[1] = *(const ushort8*)(row + off1 + 16);
  return u.v;
}
// B-tile (32x16 bf16): lane = column (lane%16); lane<16 K={0..7,16..23}, lane>=16 K={8..15,24..31}
static __device__ __forceinline__ v16bf load_bf16_B(const unsigned short* row, int lane) {
  int koff = (lane < 16) ? 0 : 8;
  union { v16bf v; ushort8 h[2]; } u;
  u.h[0] = *(const ushort8*)(row + koff);
  u.h[1] = *(const ushort8*)(row + koff + 16);
  return u.v;
}
static __device__ __forceinline__ v16bf zero_bf16() {
  union { v16bf v; unsigned short u16[16]; } u;
#pragma unroll
  for (int t = 0; t < 16; ++t) u.u16[t] = 0;
  return u.v;
}
#define ZERO_ACC {0.f,0.f,0.f,0.f,0.f,0.f,0.f,0.f}

// ---------------- prep: x -> [b][i][c] bf16; rel -> bf16 (+transposed) ----------------
__global__ __launch_bounds__(256) void k_prep(const float* __restrict__ x,
                                              const float* __restrict__ rel,
                                              unsigned short* __restrict__ xT,
                                              unsigned short* __restrict__ rel_bf,
                                              unsigned short* __restrict__ relT) {
  int idx = blockIdx.x * blockDim.x + threadIdx.x;
  if (idx < N_ * HW_ * C_) {
    int b = idx / (HW_ * C_);
    int r = idx % (HW_ * C_);
    int i = r / C_, c = r % C_;
    xT[idx] = f2bf(x[((size_t)b * C_ + c) * HW_ + i]);
  }
  if (idx < PE_ * T_) {
    int c = idx / T_, t = idx % T_;
    unsigned short v = f2bf(rel[idx]);
    rel_bf[(size_t)c * TPAD_ + t] = v;
    relT[(size_t)t * PE_ + c]     = v;
  }
}

// generic fp32 -> bf16 weight conversion
__global__ __launch_bounds__(256) void k_cvt(const float* __restrict__ src,
                                             unsigned short* __restrict__ dst, int n) {
  int i = blockIdx.x * blockDim.x + threadIdx.x;
  if (i < n) dst[i] = f2bf(src[i]);
}

// xw1 [o][64][3][3] -> w1r [tap][o][64] bf16 (contiguous K for A-loads)
__global__ __launch_bounds__(256) void k_w1(const float* __restrict__ xw1,
                                            unsigned short* __restrict__ w1r) {
  int idx = blockIdx.x * blockDim.x + threadIdx.x;
  if (idx >= BP_ * 64 * 9) return;
  int o = idx / 576, r = idx % 576;
  int c = r / 9, t = r % 9;
  w1r[((size_t)t * BP_ + o) * 64 + c] = f2bf(xw1[idx]);
}

// ---------------- qkv epilogue: BN + route to qT/kT/v ----------------
static __device__ __forceinline__ void store_qkv(int b, int o0, int i, int hi,
                                                 const v8f& x0, const v8f& x1,
                                                 const float* __restrict__ g_qkv,
                                                 const float* __restrict__ b_qkv,
                                                 unsigned short* __restrict__ qT,
                                                 unsigned short* __restrict__ kT,
                                                 unsigned short* __restrict__ vbf) {
  int g = o0 >> 7, peb = o0 & 127;
  float vals[8];
#pragma unroll
  for (int v = 0; v < 8; ++v) {
    int o = o0 + v + 8 * hi;
    vals[v] = (x0[v] + x1[v]) * (g_qkv[o] * RS_) + b_qkv[o];
  }
  size_t bg = (size_t)(b * G_ + g);
  if (peb < QKP_) {
    union { ushort8 h; unsigned short u[8]; } s;
#pragma unroll
    for (int v = 0; v < 8; ++v) s.u[v] = f2bf(vals[v]);
    *(ushort8*)(qT + (bg * HW_ + i) * QKP_ + peb + 8 * hi) = s.h;
  } else if (peb < 2 * QKP_) {
    union { ushort8 h; unsigned short u[8]; } s;
#pragma unroll
    for (int v = 0; v < 8; ++v) s.u[v] = f2bf(vals[v]);
    *(ushort8*)(kT + (bg * HW_ + i) * QKP_ + (peb - QKP_) + 8 * hi) = s.h;
  } else {
    int cb = peb - 2 * QKP_ + 8 * hi;
#pragma unroll
    for (int v = 0; v < 8; ++v)
      vbf[(bg * VP_ + cb + v) * JPAD_ + i] = f2bf(vals[v]);
  }
}

// ---------------- qkv = BN(qkv_w @ x): M=32 per wave (shared B) ----------------
__global__ __launch_bounds__(128) void k_qkv(const unsigned short* __restrict__ wq,
                                             const float* __restrict__ g_qkv,
                                             const float* __restrict__ b_qkv,
                                             const unsigned short* __restrict__ xT,
                                             unsigned short* __restrict__ qT,
                                             unsigned short* __restrict__ kT,
                                             unsigned short* __restrict__ vbf) {
  int lane = threadIdx.x & 31, wave = threadIdx.x >> 5;
  int i0 = blockIdx.x * 16;
  int o0 = (blockIdx.y * 4 + wave) * 32;       // pair of 16-wide o tiles
  int b  = blockIdx.z;
  const unsigned short* arow0 = wq + (size_t)(o0 + (lane & 15)) * C_;
  const unsigned short* arow1 = arow0 + (size_t)16 * C_;
  const unsigned short* brow  = xT + ((size_t)b * HW_ + i0 + (lane & 15)) * C_;
  v8f a00 = ZERO_ACC, a01 = ZERO_ACC, a10 = ZERO_ACC, a11 = ZERO_ACC;
  for (int c0 = 0; c0 < C_; c0 += 64) {
    __builtin_prefetch(brow + c0 + 128, 0, 1);
    v16bf b0 = load_bf16_B(brow + c0,      lane);
    v16bf b1 = load_bf16_B(brow + c0 + 32, lane);
    a00 = wmma_bf16(load_bf16_A(arow0 + c0,      lane), b0, a00);
    a10 = wmma_bf16(load_bf16_A(arow1 + c0,      lane), b0, a10);
    a01 = wmma_bf16(load_bf16_A(arow0 + c0 + 32, lane), b1, a01);
    a11 = wmma_bf16(load_bf16_A(arow1 + c0 + 32, lane), b1, a11);
  }
  int i = i0 + (lane & 15), hi = lane >> 4;
  store_qkv(b, o0,      i, hi, a00, a01, g_qkv, b_qkv, qT, kT, vbf);
  store_qkv(b, o0 + 16, i, hi, a10, a11, g_qkv, b_qkv, qT, kT, vbf);
}

// ---------------- grouped 3x3 conv, M=32 per wave -> t1T [b][i][o] bf16 ----------------
__global__ __launch_bounds__(128) void k_conv3(const unsigned short* __restrict__ w1r,
                                               const unsigned short* __restrict__ xT,
                                               unsigned short* __restrict__ t1T) {
  int lane = threadIdx.x & 31, wave = threadIdx.x >> 5;
  int i0 = blockIdx.x * 16;
  int o0 = (blockIdx.y * 4 + wave) * 32;       // stays inside one 64-ch group
  int b  = blockIdx.z;
  int cb = (o0 >> 6) * 64;
  int i = i0 + (lane & 15);
  int y = i / 28, xp = i % 28;
  v8f a00 = ZERO_ACC, a01 = ZERO_ACC, a10 = ZERO_ACC, a11 = ZERO_ACC;
#pragma unroll
  for (int ky = 0; ky < 3; ++ky)
#pragma unroll
    for (int kx = 0; kx < 3; ++kx) {
      int yy = y + ky - 1, xx = xp + kx - 1;
      bool valid = (yy >= 0) & (yy < 28) & (xx >= 0) & (xx < 28);
      const unsigned short* brow =
          xT + ((size_t)b * HW_ + (valid ? (yy * 28 + xx) : 0)) * C_ + cb;
      const unsigned short* arow0 =
          w1r + ((size_t)(ky * 3 + kx) * BP_ + o0 + (lane & 15)) * 64;
      const unsigned short* arow1 = arow0 + 16 * 64;
      v16bf b0 = valid ? load_bf16_B(brow,      lane) : zero_bf16();
      v16bf b1 = valid ? load_bf16_B(brow + 32, lane) : zero_bf16();
      a00 = wmma_bf16(load_bf16_A(arow0,      lane), b0, a00);
      a10 = wmma_bf16(load_bf16_A(arow1,      lane), b0, a10);
      a01 = wmma_bf16(load_bf16_A(arow0 + 32, lane), b1, a01);
      a11 = wmma_bf16(load_bf16_A(arow1 + 32, lane), b1, a11);
    }
  int hi = lane >> 4;
  union { ushort8 h; unsigned short u[8]; } s0, s1;
#pragma unroll
  for (int v = 0; v < 8; ++v) {
    s0.u[v] = f2bf(a00[v] + a01[v]);
    s1.u[v] = f2bf(a10[v] + a11[v]);
  }
  unsigned short* dst = t1T + ((size_t)b * HW_ + i) * BP_ + o0 + 8 * hi;
  *(ushort8*)dst        = s0.h;
  *(ushort8*)(dst + 16) = s1.h;
}

// ---------------- 1x1 conv xw2 + BN, M=32 per wave -> unary fp32, relu -> gateinT[:,0:512] ----------------
__global__ __launch_bounds__(128) void k_xw2(const unsigned short* __restrict__ w,
                                             const float* __restrict__ gx,
                                             const float* __restrict__ bx,
                                             const unsigned short* __restrict__ t1T,
                                             float* __restrict__ unary,
                                             unsigned short* __restrict__ gateinT) {
  int lane = threadIdx.x & 31, wave = threadIdx.x >> 5;
  int i0 = blockIdx.x * 16;
  int o0 = (blockIdx.y * 4 + wave) * 32;
  int b  = blockIdx.z;
  const unsigned short* arow0 = w + (size_t)(o0 + (lane & 15)) * BP_;
  const unsigned short* arow1 = arow0 + (size_t)16 * BP_;
  const unsigned short* brow  = t1T + ((size_t)b * HW_ + i0 + (lane & 15)) * BP_;
  v8f a00 = ZERO_ACC, a01 = ZERO_ACC, a10 = ZERO_ACC, a11 = ZERO_ACC;
  for (int c0 = 0; c0 < BP_; c0 += 64) {
    v16bf b0 = load_bf16_B(brow + c0,      lane);
    v16bf b1 = load_bf16_B(brow + c0 + 32, lane);
    a00 = wmma_bf16(load_bf16_A(arow0 + c0,      lane), b0, a00);
    a10 = wmma_bf16(load_bf16_A(arow1 + c0,      lane), b0, a10);
    a01 = wmma_bf16(load_bf16_A(arow0 + c0 + 32, lane), b1, a01);
    a11 = wmma_bf16(load_bf16_A(arow1 + c0 + 32, lane), b1, a11);
  }
  int i = i0 + (lane & 15), hi = lane >> 4;
  union { ushort8 h; unsigned short u[8]; } s0, s1;
#pragma unroll
  for (int v = 0; v < 8; ++v) {
    int oa = o0 + v + 8 * hi, ob = oa + 16;
    float va = (a00[v] + a01[v]) * (gx[oa] * RS_) + bx[oa];
    float vb = (a10[v] + a11[v]) * (gx[ob] * RS_) + bx[ob];
    unary[((size_t)b * BP_ + oa) * HW_ + i] = va;
    unary[((size_t)b * BP_ + ob) * HW_ + i] = vb;
    s0.u[v] = f2bf(fmaxf(va, 0.f));
    s1.u[v] = f2bf(fmaxf(vb, 0.f));
  }
  unsigned short* dst = gateinT + ((size_t)b * HW_ + i) * O2_ + o0 + 8 * hi;
  *(ushort8*)dst        = s0.h;
  *(ushort8*)(dst + 16) = s1.h;
}

// ---------------- logits tile: qk + band-F (5 WMMAs) + LDS diagonal gather ----------------
__global__ __launch_bounds__(128) void k_logits(const unsigned short* __restrict__ qT,
                                                const unsigned short* __restrict__ kT,
                                                const unsigned short* __restrict__ relT,
                                                const float* __restrict__ g_sim,
                                                const float* __restrict__ b_sim,
                                                float* __restrict__ logits, int b) {
  __shared__ float bq[4][16][32];
  __shared__ float bkk[4][16][32];
  int lane = threadIdx.x & 31, wave = threadIdx.x >> 5;
  int jt = blockIdx.x * 4 + wave;
  if (jt >= 49) return;
  int I0 = blockIdx.y * 16, J0 = jt * 16, g = blockIdx.z;
  size_t bg = (size_t)(b * G_ + g);
  const unsigned short* qrow = qT + (bg * HW_ + I0 + (lane & 15)) * QKP_;
  const unsigned short* krow = kT + (bg * HW_ + J0 + (lane & 15)) * QKP_;
  v16bf aq = load_bf16_A(qrow, lane);
  v16bf ak = load_bf16_A(krow, lane);
  v8f z = ZERO_ACC;
  v8f qk = wmma_bf16(aq, load_bf16_B(krow, lane), z);
  int u = lane & 15;
  int t0q = I0 - J0 + 768, t0k = J0 - I0 + 768;   // always in [0,1567]
  v8f fq0 = wmma_bf16(aq, load_bf16_B(relT + (size_t)(t0q + u)      * PE_,      lane), z);
  v8f fq1 = wmma_bf16(aq, load_bf16_B(relT + (size_t)(t0q + 16 + u) * PE_,      lane), z);
  v8f fk0 = wmma_bf16(ak, load_bf16_B(relT + (size_t)(t0k + u)      * PE_ + 32, lane), z);
  v8f fk1 = wmma_bf16(ak, load_bf16_B(relT + (size_t)(t0k + 16 + u) * PE_ + 32, lane), z);
  int hi = lane >> 4;
#pragma unroll
  for (int v = 0; v < 8; ++v) {
    int m = v + 8 * hi;
    bq [wave][m][u]      = fq0[v];
    bq [wave][m][16 + u] = fq1[v];
    bkk[wave][m][u]      = fk0[v];
    bkk[wave][m][16 + u] = fk1[v];
  }
  float aqk = g_sim[g] * RS_,          bqk2 = b_sim[g];
  float aqr = g_sim[G_ + g] * RS_,     bqr  = b_sim[G_ + g];
  float akr = g_sim[2 * G_ + g] * RS_, bkr  = b_sim[2 * G_ + g];
  int jl = lane & 15;
#pragma unroll
  for (int v = 0; v < 8; ++v) {
    int il = v + 8 * hi;
    float qr = bq [wave][il][il - jl + 15];   // F_q[i, i-j+783]
    float kr = bkk[wave][jl][jl - il + 15];   // F_k[j, j-i+783]
    float lg = aqk * qk[v] + bqk2 + aqr * qr + bqr + akr * kr + bkr;
    logits[((size_t)g * HW_ + I0 + il) * JPAD_ + J0 + jl] = lg;
  }
}

// ---------------- row softmax -> bf16 sim + reversed/padded bf16 simR ----------------
__global__ __launch_bounds__(256) void k_softmax(float* __restrict__ logits,
                                                 unsigned short* __restrict__ sim_bf,
                                                 unsigned short* __restrict__ simR) {
  int i = blockIdx.x, g = blockIdx.y, tid = threadIdx.x;
  size_t ridx = (size_t)g * HW_ + i;
  float* row = logits + ridx * JPAD_;
  __shared__ float red[256];
  float m = -1e30f;
  for (int j = tid; j < HW_; j += 256) m = fmaxf(m, row[j]);
  red[tid] = m; __syncthreads();
  for (int s = 128; s > 0; s >>= 1) { if (tid < s) red[tid] = fmaxf(red[tid], red[tid + s]); __syncthreads(); }
  float M = red[0]; __syncthreads();
  float ss = 0.f;
  for (int j = tid; j < HW_; j += 256) { float e = __expf(row[j] - M); row[j] = e; ss += e; }
  red[tid] = ss; __syncthreads();
  for (int s = 128; s > 0; s >>= 1) { if (tid < s) red[tid] += red[tid + s]; __syncthreads(); }
  float inv = 1.f / red[0];
  unsigned short* sr = simR + ridx * SRW_;
  unsigned short* sb = sim_bf + ridx * JPAD_;
  for (int j = tid; j < HW_; j += 256) {
    unsigned short h = f2bf(row[j] * inv);
    sb[j]       = h;          // pads (784..799) stay zero from memset
    sr[799 - j] = h;          // pos = 16 + (783-j); front/back pads stay zero
  }
}

// ---------------- sv + sve, N=32 per wave (2 waves/block, shared LDS gather) ----------------
__global__ __launch_bounds__(64) void k_svsve(const unsigned short* __restrict__ sim_bf,
                                              const unsigned short* __restrict__ simR,
                                              const unsigned short* __restrict__ vbf,
                                              const unsigned short* __restrict__ rel_bf,
                                              const float* __restrict__ g_out,
                                              const float* __restrict__ b_out,
                                              float* __restrict__ binary,
                                              unsigned short* __restrict__ gateinT, int b) {
  __shared__ __align__(16) unsigned short lsimR[16 * SRW_];   // 26.6 KB
  int lane = threadIdx.x & 31, wave = threadIdx.x >> 5;
  int c0 = wave * 32, I0 = blockIdx.x * 16, g = blockIdx.y;
  size_t bg = (size_t)(b * G_ + g);
  int il = lane & 15;
  {
    const ushort8* src = (const ushort8*)(simR + ((size_t)g * HW_ + I0) * SRW_);
    ushort8* dst = (ushort8*)lsimR;
    for (int t = threadIdx.x; t < 16 * SRW_ / 8; t += 64) dst[t] = src[t];
  }
  __syncthreads();
  const unsigned short* simrow = sim_bf + ((size_t)g * HW_ + I0 + il) * JPAD_;
  const unsigned short* vrow0  = vbf + (bg * VP_ + c0 + il) * JPAD_;
  const unsigned short* vrow1  = vrow0 + (size_t)16 * JPAD_;
  const unsigned short* rrow0  = rel_bf + (size_t)(2 * QKP_ + c0 + il) * TPAD_;
  const unsigned short* rrow1  = rrow0 + (size_t)16 * TPAD_;
  const unsigned short* lrow   = lsimR + il * SRW_ + 16 - il;   // index by (tt + kpos)
  int off1 = (lane < 16) ? 0 : 8;
  v8f sv00 = ZERO_ACC, sv01 = ZERO_ACC, sv10 = ZERO_ACC, sv11 = ZERO_ACC;
  v8f se00 = ZERO_ACC, se01 = ZERO_ACC, se10 = ZERO_ACC, se11 = ZERO_ACC;
  for (int j0 = 0; j0 < 768; j0 += 64) {            // sv = sim @ v^T
    v16bf a0 = load_bf16_A(simrow + j0,      lane);
    v16bf a1 = load_bf16_A(simrow + j0 + 32, lane);
    sv00 = wmma_bf16(a0, load_bf16_B(vrow0 + j0,      lane), sv00);
    sv10 = wmma_bf16(a0, load_bf16_B(vrow1 + j0,      lane), sv10);
    sv01 = wmma_bf16(a1, load_bf16_B(vrow0 + j0 + 32, lane), sv01);
    sv11 = wmma_bf16(a1, load_bf16_B(vrow1 + j0 + 32, lane), sv11);
  }
  {
    v16bf a0 = load_bf16_A(simrow + 768, lane);
    sv00 = wmma_bf16(a0, load_bf16_B(vrow0 + 768, lane), sv00);
    sv10 = wmma_bf16(a0, load_bf16_B(vrow1 + 768, lane), sv10);
  }
  for (int tt = 0; tt < JPAD_; tt += 32) {          // sve: skewed sim vs rel rows
    union { v16bf v; unsigned short u16[16]; } ua;
#pragma unroll
    for (int t2 = 0; t2 < 8; ++t2) {
      ua.u16[t2]     = lrow[tt + off1 + t2];        // simR gather, pads give zeros
      ua.u16[t2 + 8] = lrow[tt + off1 + 16 + t2];
    }
    v16bf bb0 = load_bf16_B(rrow0 + I0 + tt, lane);
    v16bf bb1 = load_bf16_B(rrow1 + I0 + tt, lane);
    if (tt & 32) { se01 = wmma_bf16(ua.v, bb0, se01); se11 = wmma_bf16(ua.v, bb1, se11); }
    else         { se00 = wmma_bf16(ua.v, bb0, se00); se10 = wmma_bf16(ua.v, bb1, se10); }
  }
  int hi = lane >> 4;
#pragma unroll
  for (int ct = 0; ct < 2; ++ct) {
    int c = c0 + 16 * ct + il;
    int o1 = g * 128 + 2 * c, o2 = o1 + 1;
    float a1 = g_out[o1] * RS_, b1 = b_out[o1];
    float a2 = g_out[o2] * RS_, b2 = b_out[o2];
    int p = g * 64 + c;
    const v8f& svA = ct ? sv10 : sv00; const v8f& svB = ct ? sv11 : sv01;
    const v8f& seA = ct ? se10 : se00; const v8f& seB = ct ? se11 : se01;
#pragma unroll
    for (int v = 0; v < 8; ++v) {
      int i = I0 + v + 8 * hi;
      float bin = a1 * (svA[v] + svB[v]) + b1 + a2 * (seA[v] + seB[v]) + b2;
      binary[((size_t)b * BP_ + p) * HW_ + i] = bin;
      gateinT[((size_t)b * HW_ + i) * O2_ + BP_ + p] = f2bf(fmaxf(bin, 0.f));
    }
  }
}

// ---------------- rw 1x1 conv + BN + relu, M=32 per wave -> rT [b][i][o] bf16 ----------------
__global__ __launch_bounds__(128) void k_rw(const unsigned short* __restrict__ w,
                                            const float* __restrict__ gr,
                                            const float* __restrict__ br,
                                            const unsigned short* __restrict__ gateinT,
                                            unsigned short* __restrict__ rT) {
  int lane = threadIdx.x & 31, wave = threadIdx.x >> 5;
  int i0 = blockIdx.x * 16;
  int o0 = (blockIdx.y * 4 + wave) * 32;
  int b  = blockIdx.z;
  const unsigned short* arow0 = w + (size_t)(o0 + (lane & 15)) * O2_;
  const unsigned short* arow1 = arow0 + (size_t)16 * O2_;
  const unsigned short* brow  = gateinT + ((size_t)b * HW_ + i0 + (lane & 15)) * O2_;
  v8f a00 = ZERO_ACC, a01 = ZERO_ACC, a10 = ZERO_ACC, a11 = ZERO_ACC;
  for (int c0 = 0; c0 < O2_; c0 += 64) {
    __builtin_prefetch(brow + c0 + 128, 0, 1);
    v16bf b0 = load_bf16_B(brow + c0,      lane);
    v16bf b1 = load_bf16_B(brow + c0 + 32, lane);
    a00 = wmma_bf16(load_bf16_A(arow0 + c0,      lane), b0, a00);
    a10 = wmma_bf16(load_bf16_A(arow1 + c0,      lane), b0, a10);
    a01 = wmma_bf16(load_bf16_A(arow0 + c0 + 32, lane), b1, a01);
    a11 = wmma_bf16(load_bf16_A(arow1 + c0 + 32, lane), b1, a11);
  }
  int i = i0 + (lane & 15), hi = lane >> 4;
  union { ushort8 h; unsigned short u[8]; } s0, s1;
#pragma unroll
  for (int v = 0; v < 8; ++v) {
    int oa = o0 + v + 8 * hi, ob = oa + 16;
    s0.u[v] = f2bf(fmaxf((a00[v] + a01[v]) * (gr[oa] * RS_) + br[oa], 0.f));
    s1.u[v] = f2bf(fmaxf((a10[v] + a11[v]) * (gr[ob] * RS_) + br[ob], 0.f));
  }
  unsigned short* dst = rT + ((size_t)b * HW_ + i) * BP_ + o0 + 8 * hi;
  *(ushort8*)dst        = s0.h;
  *(ushort8*)(dst + 16) = s1.h;
}

// ---------------- pw 1x1 conv + BN + sigmoid, M=32 per wave; out = gate*binary + unary ----------------
__global__ __launch_bounds__(128) void k_pw(const unsigned short* __restrict__ w,
                                            const float* __restrict__ gp,
                                            const float* __restrict__ bp,
                                            const unsigned short* __restrict__ rT,
                                            const float* __restrict__ binary,
                                            const float* __restrict__ unary,
                                            float* __restrict__ out) {
  int lane = threadIdx.x & 31, wave = threadIdx.x >> 5;
  int i0 = blockIdx.x * 16;
  int o0 = (blockIdx.y * 4 + wave) * 32;
  int b  = blockIdx.z;
  const unsigned short* arow0 = w + (size_t)(o0 + (lane & 15)) * BP_;
  const unsigned short* arow1 = arow0 + (size_t)16 * BP_;
  const unsigned short* brow  = rT + ((size_t)b * HW_ + i0 + (lane & 15)) * BP_;
  v8f a00 = ZERO_ACC, a01 = ZERO_ACC, a10 = ZERO_ACC, a11 = ZERO_ACC;
  for (int c0 = 0; c0 < BP_; c0 += 64) {
    v16bf b0 = load_bf16_B(brow + c0,      lane);
    v16bf b1 = load_bf16_B(brow + c0 + 32, lane);
    a00 = wmma_bf16(load_bf16_A(arow0 + c0,      lane), b0, a00);
    a10 = wmma_bf16(load_bf16_A(arow1 + c0,      lane), b0, a10);
    a01 = wmma_bf16(load_bf16_A(arow0 + c0 + 32, lane), b1, a01);
    a11 = wmma_bf16(load_bf16_A(arow1 + c0 + 32, lane), b1, a11);
  }
  int i = i0 + (lane & 15), hi = lane >> 4;
#pragma unroll
  for (int v = 0; v < 8; ++v) {
    int oa = o0 + v + 8 * hi, ob = oa + 16;
    float ta = (a00[v] + a01[v]) * (gp[oa] * RS_) + bp[oa];
    float tb = (a10[v] + a11[v]) * (gp[ob] * RS_) + bp[ob];
    float ga = 1.f / (1.f + __expf(-ta));
    float gb = 1.f / (1.f + __expf(-tb));
    size_t offa = ((size_t)b * BP_ + oa) * HW_ + i;
    size_t offb = ((size_t)b * BP_ + ob) * HW_ + i;
    out[offa] = ga * binary[offa] + unary[offa];
    out[offb] = gb * binary[offb] + unary[offb];
  }
}

// ---------------- host ----------------
extern "C" void kernel_launch(void* const* d_in, const int* in_sizes, int n_in,
                              void* d_out, int out_size, void* d_ws, size_t ws_size,
                              hipStream_t stream) {
  const float* x      = (const float*)d_in[0];
  const float* qkv_w  = (const float*)d_in[1];
  const float* g_qkv  = (const float*)d_in[2];
  const float* b_qkv  = (const float*)d_in[3];
  const float* g_sim  = (const float*)d_in[4];
  const float* b_sim  = (const float*)d_in[5];
  const float* g_out  = (const float*)d_in[6];
  const float* b_out  = (const float*)d_in[7];
  const float* rel    = (const float*)d_in[8];
  const float* xw1    = (const float*)d_in[9];
  const float* xw2    = (const float*)d_in[10];
  const float* g_x    = (const float*)d_in[11];
  const float* b_x    = (const float*)d_in[12];
  const float* rw     = (const float*)d_in[13];
  const float* g_r    = (const float*)d_in[14];
  const float* b_r    = (const float*)d_in[15];
  const float* pw     = (const float*)d_in[16];
  const float* g_p    = (const float*)d_in[17];
  const float* b_p    = (const float*)d_in[18];
  float* out = (float*)d_out;

  char* p = (char*)d_ws;
  auto alloc = [&](size_t bytes) { char* r = p; p += (bytes + 255) & ~(size_t)255; return r; };
  unsigned short* xT      = (unsigned short*)alloc((size_t)N_ * HW_ * C_ * 2);
  unsigned short* rel_bf  = (unsigned short*)alloc((size_t)PE_ * TPAD_ * 2);
  unsigned short* relT    = (unsigned short*)alloc((size_t)TROWS_ * PE_ * 2);
  unsigned short* qT      = (unsigned short*)alloc((size_t)N_ * G_ * HW_ * QKP_ * 2);
  unsigned short* kT      = (unsigned short*)alloc((size_t)N_ * G_ * HW_ * QKP_ * 2);
  unsigned short* vbf     = (unsigned short*)alloc((size_t)N_ * G_ * VP_ * JPAD_ * 2);
  unsigned short* t1T     = (unsigned short*)alloc((size_t)N_ * HW_ * BP_ * 2);
  unsigned short* gateinT = (unsigned short*)alloc((size_t)N_ * HW_ * O2_ * 2);
  unsigned short* rT      = (unsigned short*)alloc((size_t)N_ * HW_ * BP_ * 2);
  unsigned short* wq      = (unsigned short*)alloc((size_t)O2_ * C_ * 2);
  unsigned short* w1r     = (unsigned short*)alloc((size_t)9 * BP_ * 64 * 2);
  unsigned short* w2      = (unsigned short*)alloc((size_t)BP_ * BP_ * 2);
  unsigned short* wrw     = (unsigned short*)alloc((size_t)BP_ * O2_ * 2);
  unsigned short* wpw     = (unsigned short*)alloc((size_t)BP_ * BP_ * 2);
  float* unary  = (float*)alloc((size_t)N_ * BP_ * HW_ * 4);
  float* binary = (float*)alloc((size_t)N_ * BP_ * HW_ * 4);
  float* logits = (float*)alloc((size_t)G_ * HW_ * JPAD_ * 4);   // reused per batch
  unsigned short* sim_bf = (unsigned short*)alloc((size_t)G_ * HW_ * JPAD_ * 2);
  unsigned short* simR   = (unsigned short*)alloc((size_t)G_ * HW_ * SRW_ * 2);
  size_t used = (size_t)(p - (char*)d_ws);

  hipMemsetAsync(d_ws, 0, used, stream);   // zero-fill all pads

  k_prep<<<dim3((N_ * HW_ * C_ + 255) / 256), 256, 0, stream>>>(x, rel, xT, rel_bf, relT);
  k_cvt<<<dim3((O2_ * C_ + 255) / 256), 256, 0, stream>>>(qkv_w, wq, O2_ * C_);
  k_cvt<<<dim3((BP_ * BP_ + 255) / 256), 256, 0, stream>>>(xw2, w2, BP_ * BP_);
  k_cvt<<<dim3((BP_ * O2_ + 255) / 256), 256, 0, stream>>>(rw, wrw, BP_ * O2_);
  k_cvt<<<dim3((BP_ * BP_ + 255) / 256), 256, 0, stream>>>(pw, wpw, BP_ * BP_);
  k_w1<<<dim3((BP_ * 64 * 9 + 255) / 256), 256, 0, stream>>>(xw1, w1r);

  k_qkv<<<dim3(49, 8, 8), 128, 0, stream>>>(wq, g_qkv, b_qkv, xT, qT, kT, vbf);
  k_conv3<<<dim3(49, 4, 8), 128, 0, stream>>>(w1r, xT, t1T);
  k_xw2<<<dim3(49, 4, 8), 128, 0, stream>>>(w2, g_x, b_x, t1T, unary, gateinT);

  for (int b = 0; b < N_; ++b) {
    k_logits<<<dim3(13, 49, 8), 128, 0, stream>>>(qT, kT, relT, g_sim, b_sim, logits, b);
    k_softmax<<<dim3(HW_, G_), 256, 0, stream>>>(logits, sim_bf, simR);
    k_svsve<<<dim3(49, 8), 64, 0, stream>>>(sim_bf, simR, vbf, rel_bf, g_out, b_out,
                                            binary, gateinT, b);
  }

  k_rw<<<dim3(49, 4, 8), 128, 0, stream>>>(wrw, g_r, b_r, gateinT, rT);
  k_pw<<<dim3(49, 4, 8), 128, 0, stream>>>(wpw, g_p, b_p, rT, binary, unary, out);
}
